// CrossAttentionBlock_45689862094962
// MI455X (gfx1250) — matrix operand order repaired and checked by reading, not compiled, not run on hardware
//
#include <hip/hip_runtime.h>

// MI455X / gfx1250, wave32. Fused flash-attention cross-attention.
// Round 4: (a) V intermediate stored in 32-key tile layout [jtile][o][jloc]
// so the async global->LDS stage of V is a contiguous 8KB copy (same as K);
// (b) __launch_bounds__(128,1) gives the scheduler VGPR headroom to batch
// ds_loads per phase instead of load->wait->wmma lockstep.

typedef __attribute__((ext_vector_type(16))) _Float16 v16h;
typedef __attribute__((ext_vector_type(8)))  _Float16 v8h;
typedef __attribute__((ext_vector_type(8)))  float    v8f;

#define NPIX 6400   // 80*80
#define CHSI 128
#define CMSI 64
#define COUT 128

// ---------------- projection: q/k/v 1x1 convs, write f16 intermediates -----
// qT, kT: (B, N, 128) row-major
// v:      (B, N/32, 128, 32) tile layout == LDS tile image, linear-copyable
__global__ __launch_bounds__(256) void proj_kernel(
    const float* __restrict__ z_hsi, const float* __restrict__ z_msi,
    const float* __restrict__ Wq, const float* __restrict__ bq,
    const float* __restrict__ Wk, const float* __restrict__ bk,
    const float* __restrict__ Wv, const float* __restrict__ bv,
    _Float16* __restrict__ qT, _Float16* __restrict__ kT,
    _Float16* __restrict__ vv)
{
  __shared__ float zh[CHSI * 64];   // 32 KB
  __shared__ float zm[CMSI * 64];   // 16 KB
  const int blk = blockIdx.x;
  const int b   = blk / (NPIX / 64);
  const int i0  = (blk % (NPIX / 64)) * 64;
  const int tid = threadIdx.x;

  const float* zhb = z_hsi + (size_t)b * CHSI * NPIX;
  const float* zmb = z_msi + (size_t)b * CMSI * NPIX;
  for (int idx = tid; idx < CHSI * 64; idx += 256) {
    int c = idx >> 6, t = idx & 63;
    zh[idx] = zhb[c * NPIX + i0 + t];
  }
  for (int idx = tid; idx < CMSI * 64; idx += 256) {
    int c = idx >> 6, t = idx & 63;
    zm[idx] = zmb[c * NPIX + i0 + t];
  }
  __syncthreads();

  const int t = tid & 63;
  const int g = tid >> 6;
  const int pix = i0 + t;
  _Float16* qrow = qT + (size_t)(b * NPIX + pix) * COUT;
  _Float16* krow = kT + (size_t)(b * NPIX + pix) * COUT;
  // V tile layout: [b][pix>>5][o][pix&31]
  _Float16* vtile = vv + ((size_t)b * (NPIX / 32) + (pix >> 5)) * (COUT * 32)
                       + (pix & 31);

  for (int o = g * 32; o < g * 32 + 32; ++o) {
    float accq = bq[o];
    const float* wq = Wq + o * CHSI;
#pragma unroll 8
    for (int c = 0; c < CHSI; ++c) accq += wq[c] * zh[c * 64 + t];
    qrow[o] = (_Float16)accq;

    float acck = bk[o], accv = bv[o];
    const float* wk = Wk + o * CMSI;
    const float* wv = Wv + o * CMSI;
#pragma unroll 8
    for (int c = 0; c < CMSI; ++c) {
      float z = zm[c * 64 + t];
      acck += wk[c] * z;
      accv += wv[c] * z;
    }
    krow[o] = (_Float16)acck;
    vtile[o * 32] = (_Float16)accv;
  }
}

// ---------------- fused flash attention ------------------------------------
__global__ __launch_bounds__(128, 1) void attn_kernel(
    const _Float16* __restrict__ qT, const _Float16* __restrict__ kT,
    const _Float16* __restrict__ vv, const float* __restrict__ z_hsi,
    const float* __restrict__ gamma, float* __restrict__ out)
{
  // double-buffered K/V tiles: K tile 32x128 halfs, V tile 128x32 halfs
  __shared__ _Float16 lk[2][32 * 128];   // 2 x 8 KB
  __shared__ _Float16 lv[2][128 * 32];   // 2 x 8 KB

  const int wave  = threadIdx.x >> 5;
  const int lane  = threadIdx.x & 31;
  const int tile  = blockIdx.x * 4 + wave;          // 0 .. B*400-1
  const int b     = tile / (NPIX / 16);
  const int ibase = (tile % (NPIX / 16)) * 16;
  const int ln    = lane & 15;
  const bool hi   = lane >= 16;

  const _Float16* qTb = qT + (size_t)b * NPIX * COUT;
  const _Float16* kTb = kT + (size_t)b * NPIX * COUT;
  const _Float16* vb  = vv + (size_t)b * (NPIX / 32) * (COUT * 32);

  // async global->LDS stage: two contiguous 8KB block copies per 32-key tile.
  // LDS dest address = low 32 bits of generic pointer (AS3 offset).
  auto stage = [&](int jb, int buf) {
    const _Float16* gk = kTb + (size_t)jb * COUT;                 // 8 KB
    const _Float16* gv = vb + (size_t)(jb >> 5) * (COUT * 32);    // 8 KB
    unsigned lbk = (unsigned)(size_t)&lk[buf][0];
    unsigned lbv = (unsigned)(size_t)&lv[buf][0];
#pragma unroll
    for (int r = 0; r < 4; ++r) {
      const int f = (int)threadIdx.x + 128 * r;                   // 512 x 16B
      unsigned lk16 = lbk + (unsigned)f * 16u;
      unsigned lv16 = lbv + (unsigned)f * 16u;
      const _Float16* gak = gk + f * 8;
      const _Float16* gav = gv + f * 8;
      asm volatile("global_load_async_to_lds_b128 %0, %1, off"
                   :: "v"(lk16), "v"(gak) : "memory");
      asm volatile("global_load_async_to_lds_b128 %0, %1, off"
                   :: "v"(lv16), "v"(gav) : "memory");
    }
  };

  // B operand (fixed): Q tile; lane = col i; element e <-> K = 32c + e + 16*hi
  v16h qb[4];
  {
    const _Float16* qrow = qTb + (size_t)(ibase + ln) * COUT + (hi ? 16 : 0);
#pragma unroll
    for (int c = 0; c < 4; ++c) qb[c] = *(const v16h*)(qrow + c * 32);
  }

  v8f oacc[8];
#pragma unroll
  for (int ot = 0; ot < 8; ++ot)
#pragma unroll
    for (int e = 0; e < 8; ++e) oacc[ot][e] = 0.0f;

  float m = -1e30f, l = 0.0f;

  stage(0, 0);                                           // prologue

  for (int it = 0; it < NPIX / 32; ++it) {
    const int buf = it & 1;

    // own async copies landed; barrier => tile full AND everyone done
    // reading the buffer we are about to overwrite.
    asm volatile("s_wait_asynccnt 0x0" ::: "memory");
    __syncthreads();
    if (it * 32 + 32 < NPIX) stage(it * 32 + 32, buf ^ 1);

    const _Float16* kl = &lk[buf][0];
    const _Float16* vl = &lv[buf][0];

    // ---- load ALL K A-tiles into distinct registers ----
    const _Float16* kr0 = kl + ln * COUT;          // rows j = 0..15
    const _Float16* kr1 = kl + (16 + ln) * COUT;   // rows j = 16..31
    v16h a0[4], a1[4];
#pragma unroll
    for (int c = 0; c < 4; ++c) {
      const int kb = c * 32 + (hi ? 16 : 0);
      v8h x0 = *(const v8h*)(kr0 + kb);
      v8h y0 = *(const v8h*)(kr0 + kb + 8);
      v8h x1 = *(const v8h*)(kr1 + kb);
      v8h y1 = *(const v8h*)(kr1 + kb + 8);
#pragma unroll
      for (int e = 0; e < 8; ++e) {
        a0[c][e] = x0[e]; a0[c][e + 8] = y0[e];
        a1[c][e] = x1[e]; a1[c][e + 8] = y1[e];
      }
    }

    // ---- scores S'[j][i] = sum_o K*Q : 8 WMMAs (two accumulator chains) ----
    v8f s0, s1;
#pragma unroll
    for (int e = 0; e < 8; ++e) { s0[e] = 0.0f; s1[e] = 0.0f; }
#pragma unroll
    for (int c = 0; c < 4; ++c) {
      s0 = __builtin_amdgcn_wmma_f32_16x16x32_f16(false, a0[c], false, qb[c],
                                                  (short)0, s0, false, false);
      s1 = __builtin_amdgcn_wmma_f32_16x16x32_f16(false, a1[c], false, qb[c],
                                                  (short)0, s1, false, false);
    }

    // ---- issue V A-tile loads now; ds latency hides under softmax VALU ----
    const int jo = hi ? 16 : 0;
    v16h av[8];
#pragma unroll
    for (int ot = 0; ot < 8; ++ot) {
      const _Float16* vrow = vl + (ot * 16 + ln) * 32 + jo;
      v8h vlo = *(const v8h*)(vrow);
      v8h vhi = *(const v8h*)(vrow + 8);
#pragma unroll
      for (int e = 0; e < 8; ++e) { av[ot][e] = vlo[e]; av[ot][e + 8] = vhi[e]; }
    }

    // ---- online softmax (row i lives in lane; merge lane pair via xor 16)
    float tmax = -1e30f;
#pragma unroll
    for (int e = 0; e < 8; ++e) tmax = fmaxf(tmax, fmaxf(s0[e], s1[e]));
    tmax = fmaxf(tmax, __shfl_xor(tmax, 16));
    const float mnew  = fmaxf(m, tmax);
    const float scale = __expf(m - mnew);

    float p0[8], p1[8], rs = 0.0f;
#pragma unroll
    for (int e = 0; e < 8; ++e) {
      p0[e] = __expf(s0[e] - mnew);
      p1[e] = __expf(s1[e] - mnew);
      rs += p0[e] + p1[e];
    }
    rs += __shfl_xor(rs, 16);
    l = l * scale + rs;
    m = mnew;
#pragma unroll
    for (int ot = 0; ot < 8; ++ot)
#pragma unroll
      for (int e = 0; e < 8; ++e) oacc[ot][e] *= scale;

    // ---- assemble P^T as B operand (lane = col i, K = local key 0..31)
    float recv[8];
#pragma unroll
    for (int e = 0; e < 8; ++e) recv[e] = __shfl_xor(hi ? p0[e] : p1[e], 16);
    v16h pb;
#pragma unroll
    for (int e = 0; e < 8; ++e) {
      pb[e]     = (_Float16)(hi ? recv[e] : p0[e]);
      pb[e + 8] = (_Float16)(hi ? p1[e]   : recv[e]);
    }

    // ---- O += V(16o x 32j) x P^T(32j x 16i): 8 WMMAs ----
#pragma unroll
    for (int ot = 0; ot < 8; ++ot) {
      oacc[ot] = __builtin_amdgcn_wmma_f32_16x16x32_f16(false, av[ot], false,
                                                        pb, (short)0,
                                                        oacc[ot], false, false);
    }
  }

  // ---- finalize: out = gamma * (O / l) + z_hsi
  const float gm  = gamma[0];
  const float inv = 1.0f / l;
  const float* zb = z_hsi + (size_t)b * CHSI * NPIX;
  float*       ob = out   + (size_t)b * COUT * NPIX;
#pragma unroll
  for (int ot = 0; ot < 8; ++ot) {
#pragma unroll
    for (int e = 0; e < 8; ++e) {
      const int o = ot * 16 + e + (hi ? 8 : 0);
      const size_t off = (size_t)o * NPIX + ibase + ln;
      ob[off] = gm * (oacc[ot][e] * inv) + zb[off];
    }
  }
}

extern "C" void kernel_launch(void* const* d_in, const int* in_sizes, int n_in,
                              void* d_out, int out_size, void* d_ws, size_t ws_size,
                              hipStream_t stream) {
  const float* z_hsi = (const float*)d_in[0];
  const float* z_msi = (const float*)d_in[1];
  const float* Wq    = (const float*)d_in[2];
  const float* bq    = (const float*)d_in[3];
  const float* Wk    = (const float*)d_in[4];
  const float* bk    = (const float*)d_in[5];
  const float* Wv    = (const float*)d_in[6];
  const float* bv    = (const float*)d_in[7];
  const float* gamma = (const float*)d_in[8];
  const int B = in_sizes[0] / (CHSI * NPIX);   // = 2

  _Float16* qT = (_Float16*)d_ws;
  _Float16* kT = qT + (size_t)B * NPIX * COUT;
  _Float16* vv = kT + (size_t)B * NPIX * COUT;

  proj_kernel<<<B * (NPIX / 64), 256, 0, stream>>>(
      z_hsi, z_msi, Wq, bq, Wk, bk, Wv, bv, qT, kT, vv);

  attn_kernel<<<(B * (NPIX / 16)) / 4, 128, 0, stream>>>(
      qT, kT, vv, z_hsi, gamma, (float*)d_out);
}
